// SelfAttention_29506425323991
// MI455X (gfx1250) — matrix operand order repaired and checked by reading, not compiled, not run on hardware
//
#include <hip/hip_runtime.h>

// ---------------------------------------------------------------------------
// Causal self-attention for MI455X (gfx1250), compile-only tuning loop.
//
// x[4096,1024] f32, W[1024,3072] f32 -> out[4096,1024] f32.
// Multi-pass (192MB L2 keeps the 64MB score matrix + 32MB P resident):
//   0) xb   = bf16(x);  Wt = bf16(W)^T  (fused convert+transpose)
//   1) qvk  = x @ W            (WMMA bf16, bf16 out)   [Q | V | K]
//      Vt   = V^T              (bf16 tiled transpose)
//   2) raw  = Q @ K^T / 32     (WMMA bf16, f32 out, skip tiles above diag)
//   3) P    = causal softmax(raw)  (f32 math, bf16 out, zeros above diag)
//   4) out  = P @ V            (WMMA bf16, f32 out, K-loop clipped at diag)
//
// All GEMMs use B stored [N][K]-major so LDS staging is a straight b128 copy,
// double-buffered through the gfx1250 async global->LDS DMA path when the
// toolchain exposes it (ASYNCcnt + s_wait_asynccnt), sync b128 copy otherwise.
// ---------------------------------------------------------------------------

typedef unsigned short u16;
typedef unsigned int   u32;

typedef __attribute__((ext_vector_type(16))) __bf16 v16bf;
typedef __attribute__((ext_vector_type(8)))  float  v8f;

union FragBF { v16bf v; uint4 q[2]; };
union FragF  { v8f  v; float f[8]; };

#if __has_builtin(__builtin_amdgcn_global_load_async_to_lds_b128) && \
    __has_builtin(__builtin_amdgcn_s_wait_asynccnt)
#define USE_ASYNC_LDS 1
typedef int v4i __attribute__((vector_size(16)));       // clang builtin vector
typedef __attribute__((address_space(1))) v4i gv4i;     // global (device) AS
typedef __attribute__((address_space(3))) v4i lv4i;     // LDS (shared) AS
#else
#define USE_ASYNC_LDS 0
#endif

__device__ __forceinline__ u16 f2bf(float f) {
    u32 u = __float_as_uint(f);
    u32 r = u + 0x7FFFu + ((u >> 16) & 1u);   // round-to-nearest-even
    return (u16)(r >> 16);
}

// ---------------------------------------------------------------------------
// fp32 -> bf16 elementwise convert (for x, used row-major as A)
// ---------------------------------------------------------------------------
__global__ void cvt_f32_bf16(const float* __restrict__ src,
                             u16* __restrict__ dst, int n) {
    int idx    = blockIdx.x * blockDim.x + threadIdx.x;
    int stride = gridDim.x * blockDim.x;
    for (int i = idx; i < n; i += stride) dst[i] = f2bf(src[i]);
}

// ---------------------------------------------------------------------------
// fused convert + transpose: out[c][r] = bf16(in[r][c]); in is [R][C] f32
// ---------------------------------------------------------------------------
__global__ __launch_bounds__(256)
void cvt_transpose_f32_bf16(const float* __restrict__ in, u16* __restrict__ out,
                            int R, int C) {
    __shared__ float t[32][33];
    const int c0 = blockIdx.x * 32, r0 = blockIdx.y * 32;
    const int cx = threadIdx.x & 31, ry = threadIdx.x >> 5;   // ry: 0..7
#pragma unroll
    for (int i = 0; i < 4; ++i) {
        int r = ry + i * 8;
        t[r][cx] = in[(size_t)(r0 + r) * C + c0 + cx];
    }
    __syncthreads();
#pragma unroll
    for (int i = 0; i < 4; ++i) {
        int cc = ry + i * 8;
        out[(size_t)(c0 + cc) * R + r0 + cx] = f2bf(t[cx][cc]);
    }
}

// ---------------------------------------------------------------------------
// bf16 tiled transpose: out[c][r] = in[r][c]; in is [R][C] with row stride ldin
// ---------------------------------------------------------------------------
__global__ __launch_bounds__(256)
void transpose_bf16(const u16* __restrict__ in, u16* __restrict__ out,
                    int R, int C, int ldin) {
    __shared__ u16 t[32][34];
    const int c0 = blockIdx.x * 32, r0 = blockIdx.y * 32;
    const int cx = threadIdx.x & 31, ry = threadIdx.x >> 5;
#pragma unroll
    for (int i = 0; i < 4; ++i) {
        int r = ry + i * 8;
        t[r][cx] = in[(size_t)(r0 + r) * ldin + c0 + cx];
    }
    __syncthreads();
#pragma unroll
    for (int i = 0; i < 4; ++i) {
        int cc = ry + i * 8;
        out[(size_t)(c0 + cc) * R + r0 + cx] = t[cx][cc];
    }
}

// ---------------------------------------------------------------------------
// Tiled bf16 WMMA GEMM:  C[M,N] = scale * A[M,K] @ B^T,  B stored [N,K].
//   OUT_BF16 : store bf16, else f32
//   CSKIP    : skip output tiles strictly above the causal diagonal (pass 2)
//   CKLIM    : clip K loop at m0+BM (P is zero beyond the diagonal, pass 4)
// Workgroup tile 128x128, BK=64 (2 WMMA K-steps per staged tile), 8 waves as
// 2(M) x 4(N), each wave owns 64x32 = 4x2 WMMA accumulators.
// Double-buffered LDS; async global->LDS DMA when available.
// ---------------------------------------------------------------------------
#define BM 128
#define BN 128
#define BK 64
#define PADK 72   // ushort row stride in LDS: 144B (16B aligned, bank-skewed)

template<bool OUT_BF16, bool CSKIP, bool CKLIM>
__global__ __launch_bounds__(256)
void gemm_wmma_bf16(const u16* __restrict__ A, const u16* __restrict__ B,
                    void* __restrict__ C,
                    int M, int N, int K, int lda, int ldb, int ldc,
                    float scale) {
    __shared__ __align__(16) u16 As[2][BM * PADK];
    __shared__ __align__(16) u16 Bs[2][BN * PADK];

    const int m0 = blockIdx.y * BM;
    const int n0 = blockIdx.x * BN;
    if (CSKIP && n0 > m0 + (BM - 1)) return;   // tile fully above diagonal

    const int tid      = threadIdx.x;
    const int wave     = tid >> 5;
    const int lane     = tid & 31;
    const int waveM    = wave >> 2;    // 0..1 -> 64 rows each
    const int waveN    = wave & 3;     // 0..3 -> 32 cols each
    const int laneHalf = lane >> 4;    // 0/1
    const int lane15   = lane & 15;

    v8f acc[4][2];
#pragma unroll
    for (int mi = 0; mi < 4; ++mi)
#pragma unroll
        for (int ni = 0; ni < 2; ++ni)
            acc[mi][ni] = (v8f){0.f,0.f,0.f,0.f,0.f,0.f,0.f,0.f};

    int kEnd = K;
    if (CKLIM) { int lim = m0 + BM; kEnd = (lim < K) ? lim : K; }

    // staging: each thread moves half a 64-elem row: 64B = 4 x b128
    const int srow  = tid >> 1;        // 0..127
    const int shalf = tid & 1;         // 0/1 -> 32-elem half

    auto stage = [&](int buf, int k0) {
        const u16* gA = A + (size_t)(m0 + srow) * lda + k0 + shalf * 32;
        const u16* gB = B + (size_t)(n0 + srow) * ldb + k0 + shalf * 32;
        u16* lA = &As[buf][srow * PADK + shalf * 32];
        u16* lB = &Bs[buf][srow * PADK + shalf * 32];
#if USE_ASYNC_LDS
        gv4i* ga = (gv4i*)(u16*)gA;
        gv4i* gb = (gv4i*)(u16*)gB;
        lv4i* la = (lv4i*)lA;
        lv4i* lb = (lv4i*)lB;
        __builtin_amdgcn_global_load_async_to_lds_b128(ga, la,  0, 0);
        __builtin_amdgcn_global_load_async_to_lds_b128(ga, la, 16, 0);
        __builtin_amdgcn_global_load_async_to_lds_b128(ga, la, 32, 0);
        __builtin_amdgcn_global_load_async_to_lds_b128(ga, la, 48, 0);
        __builtin_amdgcn_global_load_async_to_lds_b128(gb, lb,  0, 0);
        __builtin_amdgcn_global_load_async_to_lds_b128(gb, lb, 16, 0);
        __builtin_amdgcn_global_load_async_to_lds_b128(gb, lb, 32, 0);
        __builtin_amdgcn_global_load_async_to_lds_b128(gb, lb, 48, 0);
#else
        const uint4* sa = (const uint4*)gA;
        const uint4* sb = (const uint4*)gB;
        uint4* da = (uint4*)lA;
        uint4* db = (uint4*)lB;
        uint4 a0 = sa[0], a1 = sa[1], a2 = sa[2], a3 = sa[3];
        uint4 b0 = sb[0], b1 = sb[1], b2 = sb[2], b3 = sb[3];
        da[0] = a0; da[1] = a1; da[2] = a2; da[3] = a3;
        db[0] = b0; db[1] = b1; db[2] = b2; db[3] = b3;
#endif
    };

    int cur = 0;
    stage(0, 0);                                   // prologue

    for (int k0 = 0; k0 < kEnd; k0 += BK) {
        const bool have_next = (k0 + BK) < kEnd;
        if (have_next) stage(cur ^ 1, k0 + BK);    // overlap next tile's DMA
#if USE_ASYNC_LDS
        if (have_next) __builtin_amdgcn_s_wait_asynccnt(8);  // tile t landed
        else           __builtin_amdgcn_s_wait_asynccnt(0);
#endif
        __syncthreads();                           // tile t visible to all waves

        const u16* Ab = As[cur];
        const u16* Bb = Bs[cur];
#pragma unroll
        for (int kw = 0; kw < BK; kw += 32) {
            // A 16x32 bf16 frag: lane=row; halves hold K {kg..kg+7, kg+16..+23}
            FragBF afrag[4];
#pragma unroll
            for (int mi = 0; mi < 4; ++mi) {
                const u16* p = Ab + (waveM * 64 + mi * 16 + lane15) * PADK
                                  + kw + laneHalf * 8;
                afrag[mi].q[0] = *(const uint4*)(p);
                afrag[mi].q[1] = *(const uint4*)(p + 16);
            }
            // B 32x16 frag: lane=col; halves hold K {kg..kg+7, kg+16..+23}
            FragBF bfrag[2];
#pragma unroll
            for (int ni = 0; ni < 2; ++ni) {
                const u16* p = Bb + (waveN * 32 + ni * 16 + lane15) * PADK
                                  + kw + laneHalf * 8;
                bfrag[ni].q[0] = *(const uint4*)(p);
                bfrag[ni].q[1] = *(const uint4*)(p + 16);
            }
#pragma unroll
            for (int mi = 0; mi < 4; ++mi)
#pragma unroll
                for (int ni = 0; ni < 2; ++ni)
                    acc[mi][ni] = __builtin_amdgcn_wmma_f32_16x16x32_bf16(
                        false, afrag[mi].v, false, bfrag[ni].v,
                        (short)0, acc[mi][ni], false, false);
        }
        __syncthreads();                           // readers done before reuse
        cur ^= 1;
    }

    // store: C 16x16 f32 layout: row = vgpr + 8*laneHalf, col = lane15
#pragma unroll
    for (int mi = 0; mi < 4; ++mi)
#pragma unroll
        for (int ni = 0; ni < 2; ++ni) {
            FragF f; f.v = acc[mi][ni];
            const int gcol = n0 + waveN * 32 + ni * 16 + lane15;
            const int grow = m0 + waveM * 64 + mi * 16 + 8 * laneHalf;
            if (OUT_BF16) {
                u16* Cb = (u16*)C;
#pragma unroll
                for (int v = 0; v < 8; ++v)
                    Cb[(size_t)(grow + v) * ldc + gcol] = f2bf(f.f[v] * scale);
            } else {
                float* Cf = (float*)C;
#pragma unroll
                for (int v = 0; v < 8; ++v)
                    Cf[(size_t)(grow + v) * ldc + gcol] = f.f[v] * scale;
            }
        }
}

// ---------------------------------------------------------------------------
// Causal softmax: one row per workgroup; bf16 probabilities, zeros above diag
// ---------------------------------------------------------------------------
__global__ __launch_bounds__(256)
void softmax_causal(const float* __restrict__ raw, u16* __restrict__ P, int S) {
    __shared__ float red[256];
    const int i   = blockIdx.x;
    const int tid = threadIdx.x;
    const int n   = i + 1;
    const float* row = raw + (size_t)i * S;
    const float L2E = 1.4426950408889634f;

    float mx = -3.402823466e38f;
    for (int j = tid; j < n; j += 256) mx = fmaxf(mx, row[j]);
    red[tid] = mx; __syncthreads();
    for (int s = 128; s > 0; s >>= 1) {
        if (tid < s) red[tid] = fmaxf(red[tid], red[tid + s]);
        __syncthreads();
    }
    mx = red[0]; __syncthreads();

    float sum = 0.f;
    for (int j = tid; j < n; j += 256) sum += exp2f((row[j] - mx) * L2E);
    red[tid] = sum; __syncthreads();
    for (int s = 128; s > 0; s >>= 1) {
        if (tid < s) red[tid] += red[tid + s];
        __syncthreads();
    }
    const float inv = 1.0f / red[0];

    u16* prow = P + (size_t)i * S;
    for (int j = tid; j < S; j += 256) {
        float p = (j < n) ? exp2f((row[j] - mx) * L2E) * inv : 0.f;
        prow[j] = f2bf(p);
    }
}

// ---------------------------------------------------------------------------
// launch
// ---------------------------------------------------------------------------
extern "C" void kernel_launch(void* const* d_in, const int* in_sizes, int n_in,
                              void* d_out, int out_size, void* d_ws, size_t ws_size,
                              hipStream_t stream) {
    constexpr int S  = 4096;   // SEQ_LEN
    constexpr int IN = 1024;   // IN_DIM
    constexpr int D3 = 3072;   // 3 * QVK_DIM
    constexpr int D  = 1024;   // QVK_DIM

    const float* x = (const float*)d_in[0];   // [S, IN]
    const float* W = (const float*)d_in[1];   // [IN, 3D]
    float* out     = (float*)d_out;           // [S, D]

    char* ws = (char*)d_ws;
    u16*   xb  = (u16*)ws;    ws += (size_t)S  * IN * 2;   // x   bf16 [S][IN]
    u16*   Wt  = (u16*)ws;    ws += (size_t)IN * D3 * 2;   // W^T bf16 [3D][IN]
    u16*   qvk = (u16*)ws;    ws += (size_t)S  * D3 * 2;   // [Q|V|K] bf16
    u16*   Vt  = (u16*)ws;    ws += (size_t)S  * D  * 2;   // V^T bf16 [D][S]
    float* raw = (float*)ws;  ws += (size_t)S  * S  * 4;   // scores f32
    u16*   P   = (u16*)ws;    ws += (size_t)S  * S  * 2;   // probs bf16
    (void)ws_size; (void)in_sizes; (void)n_in; (void)out_size;

    // 0) convert x; fused convert+transpose of W
    cvt_f32_bf16<<<2048, 256, 0, stream>>>(x, xb, S * IN);
    cvt_transpose_f32_bf16<<<dim3(D3 / 32, IN / 32), 256, 0, stream>>>(W, Wt, IN, D3);

    // 1) qvk = x @ W  (B = W^T stored [N][K]); bf16 out
    gemm_wmma_bf16<true, false, false>
        <<<dim3(D3 / BN, S / BM), 256, 0, stream>>>(
            xb, Wt, (void*)qvk, S, D3, IN, IN, IN, D3, 1.0f);

    const u16* Q  = qvk;           // cols 0..1023
    const u16* V  = qvk + D;       // cols 1024..2047 (middle chunk = vals)
    const u16* Km = qvk + 2 * D;   // cols 2048..3071 (keys)

    // V^T for pass 4 (B must be [N][K]-major)
    transpose_bf16<<<dim3(D / 32, S / 32), 256, 0, stream>>>(V, Vt, S, D, D3);

    // 2) raw = Q @ K^T / sqrt(D); skip tiles above the causal diagonal
    gemm_wmma_bf16<false, true, false>
        <<<dim3(S / BN, S / BM), 256, 0, stream>>>(
            Q, Km, (void*)raw, S, S, D, D3, D3, S, 0.03125f);

    // 3) causal softmax -> P (bf16, zeros above diagonal)
    softmax_causal<<<S, 256, 0, stream>>>(raw, P, S);

    // 4) out = P @ V (f32 out); clip K loop at the causal boundary
    gemm_wmma_bf16<false, false, true>
        <<<dim3(D / BN, S / BM), 256, 0, stream>>>(
            P, Vt, (void*)out, S, D, S, S, S, D, 1.0f);
}
